// GraphBasedMultiEncoder_43533788512745
// MI455X (gfx1250) — compile-verified, matrix operand order checked by reading, not run on hardware
//
#include <hip/hip_runtime.h>

// ---------------------------------------------------------------------------
// Types
// ---------------------------------------------------------------------------
typedef __attribute__((ext_vector_type(16))) __bf16 v16bf;
typedef __attribute__((ext_vector_type(8)))  float  v8f;
typedef unsigned int v4u __attribute__((ext_vector_type(4)));
typedef int          v8i __attribute__((ext_vector_type(8)));
typedef int          v4i __attribute__((ext_vector_type(4)));
typedef int          ivec4 __attribute__((ext_vector_type(4)));

#define LS 40  // LDS tile row stride in bf16 elements (80 B: 16B-aligned, bank-skewed)

// ---- CDNA5 feature gates (compile-safe: fall back to sync copies) ----------
#if defined(__gfx1250__) && __has_builtin(__builtin_amdgcn_global_load_async_to_lds_b128) && \
    __has_builtin(__builtin_amdgcn_s_wait_asynccnt)
#define HAVE_ASYNC 1
#else
#define HAVE_ASYNC 0
#endif

#if defined(__gfx1250__) && __has_builtin(__builtin_amdgcn_tensor_load_to_lds) && \
    __has_builtin(__builtin_amdgcn_s_wait_tensorcnt) && (__clang_major__ < 23)
#define HAVE_TDM 1
#else
#define HAVE_TDM 0
#endif

__device__ __forceinline__ unsigned short f2bf(float f) {
    unsigned int u = __builtin_bit_cast(unsigned int, f);
    u += 0x7FFFu + ((u >> 16) & 1u);          // round-to-nearest-even
    return (unsigned short)(u >> 16);
}

// Async 16-byte global -> LDS copy (ASYNCcnt); sync fallback.
// Builtin signature: (int4 AS1*, int4 AS3*, imm offset, imm cpol).
__device__ __forceinline__ void async_cp16(const void* g, void* l) {
#if HAVE_ASYNC
    __builtin_amdgcn_global_load_async_to_lds_b128(
        (__attribute__((address_space(1))) ivec4*)g,
        (__attribute__((address_space(3))) ivec4*)l, 0, 0);
#else
    *(uint4*)l = *(const uint4*)g;
#endif
}
__device__ __forceinline__ void async_wait() {
#if HAVE_ASYNC
    __builtin_amdgcn_s_wait_asynccnt(0);
#endif
}

// LDS byte offset of a __shared__ object (AS3 ptrtoint).
__device__ __forceinline__ unsigned lds_off(const void* p) {
    return (unsigned)(size_t)(__attribute__((address_space(3))) const void*)p;
}

// TDM: DMA one 64x32 bf16 tile (rows of a [N][K] weight matrix, row stride ldb
// elements) into LDS with row padding 64B->80B (pad_interval=16 dw, pad=4 dw),
// exactly matching the LS=40 fragment layout. D# per ISA 08_async_tensor §8.
__device__ __forceinline__ void tdm_load_b_tile(const unsigned short* gsrc, unsigned loff, int ldb) {
#if HAVE_TDM
    unsigned long long ga = (unsigned long long)(size_t)
        (__attribute__((address_space(1))) const void*)(const void*)gsrc;
    v4u g0 = { 1u,                                  // count=1 (valid descriptor)
               loff,                                // lds_addr [63:32]
               (unsigned)(ga & 0xFFFFFFFFu),        // global_addr [95:64]
               (unsigned)(((ga >> 32) & 0x01FFFFFFu) | (2u << 30)) }; // addr hi | type=2
    v8i g1 = { (int)((1u << 16)       // data_size = 1 (2 bytes)
                   | (1u << 20)       // pad_enable
                   | (3u << 22)       // pad_interval: 16 dwords (one 32-elem bf16 row)
                   | (3u << 25)),     // pad_amount: 4 dwords (16 B)
               (int)(32u << 16),      // tensor_dim0 = 32 (lo16 in bits [63:48])
               (int)(64u << 16),      // tensor_dim0 hi=0 | tensor_dim1 = 64
               (int)(32u << 16),      // tensor_dim1 hi=0 | tile_dim0 = 32
               (int)64,               // tile_dim1 = 64 | tile_dim2 = 0
               (int)ldb,              // tensor_dim0_stride lo32 (elements)
               0, 0 };                // stride hi | tensor_dim1_stride = 0
    v4i z = { 0, 0, 0, 0 };
    __builtin_amdgcn_tensor_load_to_lds(g0, g1, z, z, 0);
#else
    (void)gsrc; (void)loff; (void)ldb;
#endif
}

// Load one 16x32 bf16 fragment from an LDS tile (row-major [64][LS]).
// Two ds_load_b128 per lane; lanes 0-15: K 0-7/16-23, lanes 16-31: K 8-15/24-31.
__device__ __forceinline__ v16bf load_frag(const unsigned short* lds, int row0, int lane) {
    int r  = row0 + (lane & 15);
    int hi = (lane >> 4) & 1;
    const char* base = (const char*)(lds + r * LS) + hi * 16;
    union { v16bf v; uint4 q[2]; } u;
    u.q[0] = *(const uint4*)(base);
    u.q[1] = *(const uint4*)(base + 32);
    return u.v;
}

// ---------------------------------------------------------------------------
// Generic WMMA GEMM: C[M,N] = A[M,K](bf16) * B^T + bias, optional relu.
// BT_NK=true : Bsrc is [N][K] row-major (weight rows == K-major tiles; TDM path).
// BT_NK=false: Bsrc is [K][N] row-major (transpose-on-fill, sync stores).
// Block: 64x64 C tile, 256 threads (8 waves, wave = 32x16 region).
// Double-buffered LDS, async fills, one barrier per K-step.
// M,N multiples of 64; K multiple of 32 (guaranteed by problem dims).
// ---------------------------------------------------------------------------
template<bool BT_NK>
__global__ __launch_bounds__(256) void k_gemm(
    const unsigned short* __restrict__ A, int lda, long sA,
    const unsigned short* __restrict__ Bsrc, int ldb, long sB,
    const float* __restrict__ bias,
    float* __restrict__ Cf, int ldcf, long sCf, int permS, int permB,
    unsigned short* __restrict__ Cb, int ldcb, long sCb,
    int M, int N, int K, int relu)
{
    __shared__ alignas(16) unsigned short lA[2][64 * LS];
    __shared__ alignas(16) unsigned short lB[2][64 * LS];
    (void)M;
    const int bz = blockIdx.z;
    A    += (long)bz * sA;
    Bsrc += (long)bz * sB;
    const int n0 = blockIdx.x * 64, m0 = blockIdx.y * 64;
    const int tid = threadIdx.x, lane = tid & 31, w = tid >> 5;
    const int wm = w >> 2, wn = w & 3;

    auto fill = [&](int buf, int k0) {
        int rrow = tid >> 2, kseg = (tid & 3) * 8;
        async_cp16(&A[(long)(m0 + rrow) * lda + k0 + kseg], &lA[buf][rrow * LS + kseg]);
        if constexpr (BT_NK) {
#if HAVE_TDM
            if (tid < 32)  // wave 0 issues one TDM descriptor for the whole B tile
                tdm_load_b_tile(&Bsrc[(long)n0 * ldb + k0], lds_off(&lB[buf][0]), ldb);
#else
            async_cp16(&Bsrc[(long)(n0 + rrow) * ldb + k0 + kseg], &lB[buf][rrow * LS + kseg]);
#endif
        } else {
            int kk = tid >> 3, nseg = (tid & 7) * 8;   // lB[n][k] = Bsrc[k][n]
            uint4 q = *(const uint4*)&Bsrc[(long)(k0 + kk) * ldb + n0 + nseg];
            const unsigned short* e = (const unsigned short*)&q;
            #pragma unroll
            for (int j = 0; j < 8; ++j) lB[buf][(nseg + j) * LS + kk] = e[j];
        }
    };
    auto wait_fill = [&]() {
        async_wait();
#if HAVE_TDM
        if (BT_NK && tid < 32) __builtin_amdgcn_s_wait_tensorcnt(0);
#endif
    };

    v8f zero = {};
    v8f acc0 = zero, acc1 = zero;

    fill(0, 0);
    wait_fill();
    __syncthreads();
    int cur = 0;
    for (int k0 = 0; k0 < K; k0 += 32, cur ^= 1) {
        if (k0 + 32 < K) fill(cur ^ 1, k0 + 32);   // prefetch next tiles into other buffer
        v16bf b  = load_frag(lB[cur], wn * 16, lane);
        v16bf a0 = load_frag(lA[cur], wm * 32, lane);
        v16bf a1 = load_frag(lA[cur], wm * 32 + 16, lane);
        acc0 = __builtin_amdgcn_wmma_f32_16x16x32_bf16(false, a0, false, b, (short)0, acc0, false, false);
        acc1 = __builtin_amdgcn_wmma_f32_16x16x32_bf16(false, a1, false, b, (short)0, acc1, false, false);
        wait_fill();
        __syncthreads();
    }

    const int col = n0 + wn * 16 + (lane & 15);
    const float bv = bias ? bias[col] : 0.f;
    const int hi8 = ((lane >> 4) & 1) * 8;
    float* cf = Cf ? Cf + (long)bz * sCf : nullptr;
    unsigned short* cb = Cb ? Cb + (long)bz * sCb : nullptr;
    #pragma unroll
    for (int s2 = 0; s2 < 2; ++s2) {
        v8f a = s2 ? acc1 : acc0;
        #pragma unroll
        for (int v = 0; v < 8; ++v) {
            int row = m0 + wm * 32 + s2 * 16 + hi8 + v;
            float x = a[v] + bv;
            if (relu) x = fmaxf(x, 0.f);
            if (cf) {
                long idx;
                if (permS > 0) { int b_ = row / permS, s_ = row % permS;
                                 idx = ((long)s_ * permB + b_) * ldcf + col; }
                else idx = (long)row * ldcf + col;
                cf[idx] = x;
            }
            if (cb) cb[(long)row * ldcb + col] = f2bf(x);
        }
    }
}

// ---------------------------------------------------------------------------
// Fused GRU step: hg = h_prev @ Whh^T (3 gates via WMMA) then gate math.
// grid: (512/64 cols, 128/64 rows, 2 dirs); 256 threads. Double-buffered async.
// ---------------------------------------------------------------------------
__global__ __launch_bounds__(256) void k_gru_step(
    int t, int last,
    const float* __restrict__ xg,            // [2][32768*1536] (b_ih included)
    const unsigned short* __restrict__ hp_b, // prev slot [2][128*512]
    const float* __restrict__ hp_f,
    unsigned short* __restrict__ hn_b,       // next slot
    float* __restrict__ hn_f,
    const unsigned short* __restrict__ whh,  // [2][1536*512]
    const float* __restrict__ bhhF, const float* __restrict__ bhhB,
    float* __restrict__ xnext,               // [32768][1024]
    float* __restrict__ hid)                 // layer base, +d*128*512
{
    __shared__ alignas(16) unsigned short lA[2][64 * LS];
    __shared__ alignas(16) unsigned short lB[2][3][64 * LS];
    const int d = blockIdx.z;
    const int time = d ? (255 - t) : t;
    const unsigned short* hp = hp_b + d * (128 * 512);
    const float* hpf = hp_f + d * (128 * 512);
    const unsigned short* W = whh + (long)d * 1536 * 512;
    const float* bhh = d ? bhhB : bhhF;
    const float* xgd = xg + (long)d * ((long)32768 * 1536) + (long)time * 128 * 1536;
    const int m0 = blockIdx.y * 64, n0 = blockIdx.x * 64;
    const int tid = threadIdx.x, lane = tid & 31, w = tid >> 5;
    const int wm = w >> 2, wn = w & 3;

    auto fill = [&](int buf, int k0) {
        #pragma unroll
        for (int i = 0; i < 4; ++i) {  // 4 tiles (A + 3 gate B tiles), 1 b128 each
            int idx = tid + i * 256;
            int tile = idx >> 8, u = idx & 255;
            int row = u >> 2, kseg = (u & 3) * 8;
            const unsigned short* src = (tile == 0)
                ? &hp[(m0 + row) * 512 + k0 + kseg]
                : &W[((tile - 1) * 512 + n0 + row) * 512 + k0 + kseg];
            unsigned short* dst = (tile == 0) ? &lA[buf][row * LS + kseg]
                                              : &lB[buf][tile - 1][row * LS + kseg];
            async_cp16(src, dst);
        }
    };

    v8f zero = {};
    v8f acc[3][2];
    #pragma unroll
    for (int g = 0; g < 3; ++g) { acc[g][0] = zero; acc[g][1] = zero; }

    fill(0, 0);
    async_wait();
    __syncthreads();
    int cur = 0;
    for (int k0 = 0; k0 < 512; k0 += 32, cur ^= 1) {
        if (k0 + 32 < 512) fill(cur ^ 1, k0 + 32);
        v16bf a0 = load_frag(lA[cur], wm * 32, lane);
        v16bf a1 = load_frag(lA[cur], wm * 32 + 16, lane);
        #pragma unroll
        for (int g = 0; g < 3; ++g) {
            v16bf bb = load_frag(&lB[cur][g][0], wn * 16, lane);
            acc[g][0] = __builtin_amdgcn_wmma_f32_16x16x32_bf16(false, a0, false, bb, (short)0, acc[g][0], false, false);
            acc[g][1] = __builtin_amdgcn_wmma_f32_16x16x32_bf16(false, a1, false, bb, (short)0, acc[g][1], false, false);
        }
        async_wait();
        __syncthreads();
    }

    const int col = n0 + wn * 16 + (lane & 15);
    const float br = bhh[col], bz = bhh[512 + col], bn = bhh[1024 + col];
    const int hi8 = ((lane >> 4) & 1) * 8;
    #pragma unroll
    for (int s2 = 0; s2 < 2; ++s2) {
        #pragma unroll
        for (int v = 0; v < 8; ++v) {
            int row = m0 + wm * 32 + s2 * 16 + hi8 + v;
            long gx = (long)row * 1536;
            float xr = xgd[gx + col], xz = xgd[gx + 512 + col], xn = xgd[gx + 1024 + col];
            float r = 1.f / (1.f + __expf(-(xr + acc[0][s2][v] + br)));
            float z = 1.f / (1.f + __expf(-(xz + acc[1][s2][v] + bz)));
            float n = tanhf(xn + r * (acc[2][s2][v] + bn));
            float h = (1.f - z) * n + z * hpf[row * 512 + col];
            long hidx = (long)d * (128 * 512) + row * 512 + col;
            hn_f[hidx] = h;
            hn_b[hidx] = f2bf(h);
            xnext[((long)time * 128 + row) * 1024 + d * 512 + col] = h;
            if (last) hid[hidx] = h;
        }
    }
}

// ---------------------------------------------------------------------------
// Elementwise helpers
// ---------------------------------------------------------------------------
__global__ void k_convert(const float* __restrict__ s, unsigned short* __restrict__ d, long n) {
    long i = (long)blockIdx.x * blockDim.x + threadIdx.x;
    if (i < n) d[i] = f2bf(s[i]);
}

__global__ void k_embed(const int* __restrict__ id1, const int* __restrict__ id2,
                        const float* __restrict__ e1, const float* __restrict__ e2,
                        unsigned short* __restrict__ x0) {
    long i = (long)blockIdx.x * blockDim.x + threadIdx.x;
    if (i >= (long)32768 * 192) return;
    int f = (int)(i % 192); long tok = i / 192;
    float v;
    if (f < 128) { int id = id1[tok]; v = e1[(long)id * 128 + f]; }
    else         { int id = id2[tok]; v = (id == 0) ? 0.f : e2[(long)id * 64 + (f - 128)]; }
    x0[i] = f2bf(v);
}

__global__ void k_dinv(const int* __restrict__ pg, float* __restrict__ dv) {
    int i = blockIdx.x * blockDim.x + threadIdx.x;
    if (i >= 128 * 256) return;
    int b = i >> 8, s = i & 255;
    const int* row = pg + ((long)b * 3 + 2) * 65536 + s * 256;
    int sum = 1;  // self-loop
    for (int t2 = 0; t2 < 256; ++t2) sum += row[t2];
    dv[i] = rsqrtf((float)sum);
}

__global__ void k_adj(const int* __restrict__ pg, const float* __restrict__ dv,
                      unsigned short* __restrict__ An) {
    long i = (long)blockIdx.x * blockDim.x + threadIdx.x;
    if (i >= (long)128 * 65536) return;
    int c = (int)(i & 255); long r_ = i >> 8; int s = (int)(r_ & 255); int b = (int)(r_ >> 8);
    float a = (float)pg[((long)b * 3 + 2) * 65536 + s * 256 + c] + (s == c ? 1.f : 0.f);
    An[i] = f2bf(a * dv[b * 256 + s] * dv[b * 256 + c]);
}

__global__ void k_sumdir(const float* __restrict__ x, unsigned short* __restrict__ nb) {
    long i = (long)blockIdx.x * blockDim.x + threadIdx.x;
    if (i >= (long)32768 * 512) return;
    int h = (int)(i & 511); long bs = i >> 9; int s = (int)(bs & 255); int b = (int)(bs >> 8);
    long src = ((long)s * 128 + b) * 1024 + h;
    nb[i] = f2bf(x[src] + x[src + 512]);
}

__global__ void k_concat(const unsigned short* __restrict__ a, const unsigned short* __restrict__ b,
                         unsigned short* __restrict__ c) {
    long i = (long)blockIdx.x * blockDim.x + threadIdx.x;
    if (i >= (long)32768 * 1024) return;
    int cc = (int)(i & 1023); long r = i >> 10;
    c[i] = (cc < 512) ? a[r * 512 + cc] : b[r * 512 + cc - 512];
}

// ---------------------------------------------------------------------------
// Host orchestration
// ---------------------------------------------------------------------------
extern "C" void kernel_launch(void* const* d_in, const int* in_sizes, int n_in,
                              void* d_out, int out_size, void* d_ws, size_t ws_size,
                              hipStream_t stream) {
    (void)in_sizes; (void)n_in; (void)out_size; (void)ws_size;
    constexpr long NT = 32768;          // S*B tokens
    const int* in1 = (const int*)d_in[0];
    const int* in2 = (const int*)d_in[1];
    const int* pg  = (const int*)d_in[3];
    const float* emb1 = (const float*)d_in[4];
    const float* emb2 = (const float*)d_in[5];
    const float* wih[2][2] = {{(const float*)d_in[6],  (const float*)d_in[10]},
                              {(const float*)d_in[14], (const float*)d_in[18]}};
    const float* whh[2][2] = {{(const float*)d_in[7],  (const float*)d_in[11]},
                              {(const float*)d_in[15], (const float*)d_in[19]}};
    const float* bih[2][2] = {{(const float*)d_in[8],  (const float*)d_in[12]},
                              {(const float*)d_in[16], (const float*)d_in[20]}};
    const float* bhh[2][2] = {{(const float*)d_in[9],  (const float*)d_in[13]},
                              {(const float*)d_in[17], (const float*)d_in[21]}};
    const float* fc1w = (const float*)d_in[22]; const float* fc1b = (const float*)d_in[23];
    const float* fc2w = (const float*)d_in[24]; const float* fc2b = (const float*)d_in[25];
    const float* outw = (const float*)d_in[26]; const float* outb = (const float*)d_in[27];
    float* out_f = (float*)d_out;
    float* hid_base = out_f + (long)256 * 128 * 512;

    char* ws = (char*)d_ws;
    size_t off = 0;
    auto carve = [&](size_t bytes) -> char* {
        char* p = ws + off; off = (off + bytes + 255) & ~(size_t)255; return p;
    };
    unsigned short* x0b   = (unsigned short*)carve(NT * 192 * 2);
    unsigned short* x1b   = (unsigned short*)carve(NT * 1024 * 2);
    unsigned short* wihb  = (unsigned short*)carve((size_t)2 * 1536 * 1024 * 2);
    unsigned short* whhb  = (unsigned short*)carve((size_t)2 * 1536 * 512 * 2);
    float*          xg    = (float*)carve((size_t)2 * NT * 1536 * 4);
    float*          hf    = (float*)carve((size_t)2 * 2 * 128 * 512 * 4);  // [slot][dir]
    unsigned short* hb    = (unsigned short*)carve((size_t)2 * 2 * 128 * 512 * 2);
    float*          xnext = (float*)carve(NT * 1024 * 4);
    unsigned short* nodeb = (unsigned short*)carve(NT * 512 * 2);
    unsigned short* nib   = (unsigned short*)carve(NT * 512 * 2);
    unsigned short* tmpb  = (unsigned short*)carve(NT * 512 * 2);
    unsigned short* catb  = (unsigned short*)carve(NT * 1024 * 2);
    unsigned short* Anb   = (unsigned short*)carve((size_t)128 * 65536 * 2);
    float*          dinv  = (float*)carve(128 * 256 * 4);
    unsigned short* gw1   = (unsigned short*)carve((size_t)2 * 512 * 512 * 2);
    unsigned short* gw2   = (unsigned short*)carve((size_t)2 * 512 * 512 * 2);
    unsigned short* gwo   = (unsigned short*)carve((size_t)2 * 512 * 1024 * 2);

    auto cvt = [&](const float* s, unsigned short* d, long n) {
        k_convert<<<(unsigned)((n + 255) / 256), 256, 0, stream>>>(s, d, n);
    };

    // embeddings -> bf16 [S*B, 192]
    k_embed<<<(unsigned)((NT * 192 + 255) / 256), 256, 0, stream>>>(in1, in2, emb1, emb2, x0b);

    const long HSLOT = (long)2 * 128 * 512;  // elems per h slot (both dirs)

    for (int l = 0; l < 2; ++l) {
        const int Kin = (l == 0) ? 192 : 1024;
        const unsigned short* xa = (l == 0) ? x0b : x1b;
        for (int d = 0; d < 2; ++d) {
            cvt(wih[l][d], wihb + (size_t)d * 1536 * 1024, (long)1536 * Kin);
            cvt(whh[l][d], whhb + (size_t)d * 1536 * 512, (long)1536 * 512);
        }
        for (int d = 0; d < 2; ++d) {
            // xg[d] = x @ Wih^T + b_ih   (M=32768, N=1536, K=Kin)
            k_gemm<true><<<dim3(1536 / 64, (unsigned)(NT / 64), 1), 256, 0, stream>>>(
                xa, Kin, 0, wihb + (size_t)d * 1536 * 1024, Kin, 0, bih[l][d],
                xg + (size_t)d * NT * 1536, 1536, 0, 0, 0,
                nullptr, 0, 0, (int)NT, 1536, Kin, 0);
        }
        (void)hipMemsetAsync(hf, 0, (size_t)HSLOT * 4, stream);   // slot 0 = h0 = 0
        (void)hipMemsetAsync(hb, 0, (size_t)HSLOT * 2, stream);
        for (int t = 0; t < 256; ++t) {
            size_t ps = (size_t)(t & 1) * HSLOT, ns = (size_t)((t + 1) & 1) * HSLOT;
            k_gru_step<<<dim3(8, 2, 2), 256, 0, stream>>>(
                t, (t == 255) ? 1 : 0, xg,
                hb + ps, hf + ps, hb + ns, hf + ns,
                whhb, bhh[l][0], bhh[l][1],
                xnext, hid_base + (size_t)l * 2 * 128 * 512);
        }
        if (l == 0) cvt(xnext, x1b, NT * 1024);  // next layer input in bf16
    }

    // ---- GNN ----
    k_sumdir<<<(unsigned)((NT * 512 + 255) / 256), 256, 0, stream>>>(xnext, nodeb);
    k_dinv<<<(unsigned)((128 * 256 + 255) / 256), 256, 0, stream>>>(pg, dinv);
    k_adj<<<(unsigned)(((long)128 * 65536 + 255) / 256), 256, 0, stream>>>(pg, dinv, Anb);
    cvt(fc1w, gw1, (long)2 * 512 * 512);
    cvt(fc2w, gw2, (long)2 * 512 * 512);
    cvt(outw, gwo, (long)2 * 512 * 1024);

    for (int i = 0; i < 2; ++i) {
        // tmp = An @ node   (batched over 128; B source is [K=t][N=h] -> transpose fill)
        k_gemm<false><<<dim3(8, 4, 128), 256, 0, stream>>>(
            Anb, 256, 65536, nodeb, 512, 131072, nullptr,
            nullptr, 0, 0, 0, 0, tmpb, 512, 131072, 256, 512, 256, 0);
        // ni = relu(tmp @ fc1^T + b)
        k_gemm<true><<<dim3(8, (unsigned)(NT / 64), 1), 256, 0, stream>>>(
            tmpb, 512, 0, gw1 + (size_t)i * 512 * 512, 512, 0, fc1b + i * 512,
            nullptr, 0, 0, 0, 0, nib, 512, 0, (int)NT, 512, 512, 1);
        // tmp = An @ ni
        k_gemm<false><<<dim3(8, 4, 128), 256, 0, stream>>>(
            Anb, 256, 65536, nib, 512, 131072, nullptr,
            nullptr, 0, 0, 0, 0, tmpb, 512, 131072, 256, 512, 256, 0);
        // ni = relu(tmp @ fc2^T + b)
        k_gemm<true><<<dim3(8, (unsigned)(NT / 64), 1), 256, 0, stream>>>(
            tmpb, 512, 0, gw2 + (size_t)i * 512 * 512, 512, 0, fc2b + i * 512,
            nullptr, 0, 0, 0, 0, nib, 512, 0, (int)NT, 512, 512, 1);
        // cat = [node | ni]
        k_concat<<<(unsigned)((NT * 1024 + 255) / 256), 256, 0, stream>>>(nodeb, nib, catb);
        // node = relu(cat @ out_w^T + b); last hop writes f32 permuted to [S,B,H]
        if (i == 0) {
            k_gemm<true><<<dim3(8, (unsigned)(NT / 64), 1), 256, 0, stream>>>(
                catb, 1024, 0, gwo + (size_t)i * 512 * 1024, 1024, 0, outb + i * 512,
                nullptr, 0, 0, 0, 0, nodeb, 512, 0, (int)NT, 512, 1024, 1);
        } else {
            k_gemm<true><<<dim3(8, (unsigned)(NT / 64), 1), 256, 0, stream>>>(
                catb, 1024, 0, gwo + (size_t)i * 512 * 1024, 1024, 0, outb + i * 512,
                out_f, 512, 0, 256, 128, nullptr, 0, 0, (int)NT, 512, 1024, 1);
        }
    }
}
